// RNNVQVAE_65395172049333
// MI455X (gfx1250) — matrix operand order repaired and checked by reading, not compiled, not run on hardware
//
#include <hip/hip_runtime.h>
#include <hip/hip_bf16.h>
#include <math.h>

typedef __attribute__((ext_vector_type(16))) _Float16 v16h;
typedef __attribute__((ext_vector_type(8)))  float    v8f;
typedef __attribute__((ext_vector_type(8)))  unsigned v8u;

#define BB   128
#define TTS  512
#define IND  32
#define HH   256
#define DD   512
#define NE   1024
#define BETA_C 0.25f

// workspace layout in u32 units
#define WS_WH   0                    // 2 dirs * 8kt*48nt*32lane*8vgpr = 2*98304
#define WS_WX   (WS_WH + 2*98304)    // 2 dirs * 48nt*32lane*8vgpr   = 2*12288
#define WS_ZEL  (WS_WX + 2*12288)    // 128*512 f32
#define WS_IDX  (WS_ZEL + 65536)     // 128 int
#define WS_PART (WS_IDX + 128)       // 128 f32

union FragU { v8u u; v16h h; };

static __device__ __forceinline__ v8f wmma_f16(v16h a, v16h b, v8f c) {
  return __builtin_amdgcn_wmma_f32_16x16x32_f16(false, a, false, b, (short)0, c,
                                                false, false);
}
static __device__ __forceinline__ unsigned pkh2(float a, float b) {
  union { _Float16 h[2]; unsigned u; } x;
  x.h[0] = (_Float16)a; x.h[1] = (_Float16)b; return x.u;
}
static __device__ __forceinline__ float sigmoidf_(float x) {
  return 1.0f / (1.0f + __expf(-x));
}
static __device__ __forceinline__ v8f zero8() {
  v8f z = {0.f,0.f,0.f,0.f,0.f,0.f,0.f,0.f}; return z;
}

// ---------------------------------------------------------------------------
// Pack Wh / Wx (f32 row-major [N=768][K]) into WMMA B-fragment order:
// B[k][n] = W[n][k];  lane = (n&15) + 16*(k>>4);  vgpr = (k&15)>>1;  half = k&1
// flat: ((kt*48 + nt)*32 + lane)*8 + vgpr   (kt: K/32 chunks)
// ---------------------------------------------------------------------------
__global__ __launch_bounds__(256) void pack_weights(const float* Wx_f, const float* Wh_f,
                                                    const float* Wx_b, const float* Wh_b,
                                                    unsigned* ws) {
  int gid = blockIdx.x * 256 + threadIdx.x;
  const int per_dir = 98304 + 12288;
  if (gid >= 2 * per_dir) return;
  int dir = gid / per_dir;
  int rem = gid - dir * per_dir;
  const float* Wx = dir ? Wx_b : Wx_f;
  const float* Wh = dir ? Wh_b : Wh_f;
  if (rem < 98304) {            // Wh: K=256 -> 8 kt
    int v = rem & 7, t = rem >> 3;
    int lane = t & 31; t >>= 5;
    int nt = t % 48, kt = t / 48;
    int kl = ((lane >> 4) << 4) + (v << 1);
    int ng = nt * 16 + (lane & 15);
    int kg = kt * 32 + kl;
    ws[WS_WH + dir * 98304 + rem] = pkh2(Wh[ng * 256 + kg], Wh[ng * 256 + kg + 1]);
  } else {                      // Wx: K=32 -> 1 kt
    int r2 = rem - 98304;
    int v = r2 & 7, t = r2 >> 3;
    int lane = t & 31; t >>= 5;
    int nt = t;
    int kl = ((lane >> 4) << 4) + (v << 1);
    int ng = nt * 16 + (lane & 15);
    ws[WS_WX + dir * 12288 + r2] = pkh2(Wx[ng * 32 + kl], Wx[ng * 32 + kl + 1]);
  }
}

// ---------------------------------------------------------------------------
// Persistent GRU kernel: 16 blocks = {2 dirs} x {8 batch tiles of M=16}.
// No inter-workgroup sync: each WG owns its 16 batch rows for all 512 steps.
// ---------------------------------------------------------------------------
__global__ __launch_bounds__(256) void gru_kernel(const float* __restrict__ traj,
                                                  const int*   __restrict__ mask,
                                                  const float* bx_f, const float* bh_f,
                                                  const float* bx_b, const float* bh_b,
                                                  unsigned* ws) {
  __shared__ __align__(32) unsigned ax_pack[256];     // x_t A-fragments (16x32 f16)
  __shared__ __align__(32) unsigned ah_pack[2048];    // h   A-fragments (8 kt x 16x32 f16)
  __shared__ float h_cur[16 * 256];
  __shared__ int   lidx[16];

  const int tid  = threadIdx.x;
  const int lane = tid & 31;
  const int w    = tid >> 5;              // wave id 0..7
  const int dir  = blockIdx.x >> 3;       // 0 = fwd, 1 = bwd
  const int b0   = (blockIdx.x & 7) * 16;

  const float* bx = dir ? bx_b : bx_f;
  const float* bh = dir ? bh_b : bh_f;
  const v8u* pWh = (const v8u*)(ws + WS_WH + dir * 98304);
  const v8u* pWx = (const v8u*)(ws + WS_WX + dir * 12288);
  float* zel = (float*)(ws + WS_ZEL);

  for (int i = tid; i < 16 * 256; i += 256) h_cur[i] = 0.0f;
  if (tid < 16) {                         // last index where mask==1
    const int* mr = mask + (size_t)(b0 + tid) * TTS;
    int last = 0;
    for (int t = 0; t < TTS; ++t) if (mr[t]) last = t;
    lidx[tid] = last;
  }
  __syncthreads();

  const int m_base = (lane >> 4) * 8;     // C/D layout: m = v + 8*(lane>>4)

  // loop-invariant per-wave: column bases, fused biases, Wx B-fragments
  v16h  bxf[2][3];
  float br_[2], bz_[2], bnx_[2], bnh_[2];
  int   jcol[2];
  for (int g = 0; g < 2; ++g) {
    int nt_r = w + 8 * g;                 // r-gate n-tile (0..15)
    jcol[g] = nt_r * 16 + (lane & 15);    // hidden column owned by this lane
    int j = jcol[g];
    br_[g]  = bx[j]          + bh[j];
    bz_[g]  = bx[HH + j]     + bh[HH + j];
    bnx_[g] = bx[2 * HH + j];
    bnh_[g] = bh[2 * HH + j];
    for (int q = 0; q < 3; ++q) {         // r, z, n tiles: nt_r + 16*q
      FragU f; f.u = pWx[(nt_r + 16 * q) * 32 + lane];
      bxf[g][q] = f.h;
    }
  }

  for (int s = 0; s < TTS; ++s) {
    const int t = dir ? (TTS - 1 - s) : s;

    // --- pack x_t into A-fragment layout (1 u32 per thread) ---
    {
      int le = tid >> 3, ve = tid & 7;
      int m = le & 15, g = le >> 4;
      int kl = 16 * g + 2 * ve;
      const float* src = traj + ((size_t)(b0 + m) * TTS + t) * IND + kl;
      ax_pack[tid] = pkh2(src[0], src[1]);
    }
    // --- pack h into A-fragment layout (8 u32 per thread) ---
    {
      int kt = tid >> 5, le = tid & 31;
      int m = le & 15, g = le >> 4;
#pragma unroll
      for (int v = 0; v < 8; ++v) {
        int kl = 16 * g + 2 * v;
        int kg = kt * 32 + kl;
        ah_pack[tid * 8 + v] = pkh2(h_cur[m * 256 + kg], h_cur[m * 256 + kg + 1]);
      }
    }
    __syncthreads();

    FragU axf; axf.u = *(const v8u*)&ax_pack[lane * 8];
    v16h ah[8];
#pragma unroll
    for (int kt = 0; kt < 8; ++kt) {
      FragU f; f.u = *(const v8u*)&ah_pack[kt * 256 + lane * 8];
      ah[kt] = f.h;
    }

#pragma unroll
    for (int g = 0; g < 2; ++g) {
      const int nt_r = w + 8 * g;
      v8f cxr = zero8(), cxz = zero8(), cxn = zero8();
      v8f chr_ = zero8(), chz_ = zero8(), chn_ = zero8();
      // x-gate contributions (K=32, one WMMA each)
      cxr = wmma_f16(axf.h, bxf[g][0], cxr);
      cxz = wmma_f16(axf.h, bxf[g][1], cxz);
      cxn = wmma_f16(axf.h, bxf[g][2], cxn);
      // h-gate contributions (K=256 -> 8 WMMAs each), Wh streamed from L2
#pragma unroll
      for (int kt = 0; kt < 8; ++kt) {
        FragU fr, fz, fn;
        fr.u = pWh[(kt * 48 + nt_r)      * 32 + lane];
        fz.u = pWh[(kt * 48 + nt_r + 16) * 32 + lane];
        fn.u = pWh[(kt * 48 + nt_r + 32) * 32 + lane];
        chr_ = wmma_f16(ah[kt], fr.h, chr_);
        chz_ = wmma_f16(ah[kt], fz.h, chz_);
        chn_ = wmma_f16(ah[kt], fn.h, chn_);
      }
      // gate math fully in-registers (wave owns matching r/z/n columns)
      const int j = jcol[g];
#pragma unroll
      for (int v = 0; v < 8; ++v) {
        int   m    = v + m_base;
        float r    = sigmoidf_(cxr[v] + chr_[v] + br_[g]);
        float z    = sigmoidf_(cxz[v] + chz_[v] + bz_[g]);
        float n    = tanhf(cxn[v] + bnx_[g] + r * (chn_[v] + bnh_[g]));
        float hold = h_cur[m * 256 + j];
        float hnew = (1.0f - z) * n + z * hold;
        h_cur[m * 256 + j] = hnew;
        if (t == lidx[m])
          zel[(size_t)(b0 + m) * DD + dir * HH + j] = hnew;
      }
    }
    __syncthreads();
  }
}

// ---------------------------------------------------------------------------
// VQ: one block per batch row; argmin over |e|^2 - 2 z.e (|z|^2 constant).
// ---------------------------------------------------------------------------
__global__ __launch_bounds__(256) void vq_kernel(const float* __restrict__ cb,
                                                 unsigned* ws, float* out) {
  __shared__ float zrow[512];
  __shared__ float red[256];
  __shared__ int   redi[256];
  __shared__ int   s_best;
  const float* zel = (const float*)(ws + WS_ZEL);
  const int b = blockIdx.x, tid = threadIdx.x;
  zrow[tid]       = zel[(size_t)b * 512 + tid];
  zrow[tid + 256] = zel[(size_t)b * 512 + tid + 256];
  __syncthreads();

  float bd = 3.4e38f; int bi = NE;
  for (int c = tid; c < NE; c += 256) {
    const float* row = cb + (size_t)c * 512;
    float dot = 0.f, nrm = 0.f;
    for (int k = 0; k < 512; ++k) { float wv = row[k]; dot += wv * zrow[k]; nrm += wv * wv; }
    float d = nrm - 2.0f * dot;
    if (d < bd) { bd = d; bi = c; }       // c ascending -> keeps first min
  }
  red[tid] = bd; redi[tid] = bi;
  __syncthreads();
  for (int sn = 128; sn > 0; sn >>= 1) {
    if (tid < sn) {
      float od = red[tid + sn]; int oi = redi[tid + sn];
      if (od < red[tid] || (od == red[tid] && oi < redi[tid])) { red[tid] = od; redi[tid] = oi; }
    }
    __syncthreads();
  }
  if (tid == 0) s_best = redi[0];
  __syncthreads();
  const int best = s_best;

  float psum = 0.f;
  for (int k = tid; k < 512; k += 256) {
    float q = cb[(size_t)best * 512 + k];
    out[(size_t)b * 512 + k] = q;         // z_q (straight-through == z_q in value)
    float dfv = q - zrow[k];
    psum += dfv * dfv;
  }
  red[tid] = psum;
  __syncthreads();
  for (int sn = 128; sn > 0; sn >>= 1) { if (tid < sn) red[tid] += red[tid + sn]; __syncthreads(); }
  if (tid == 0) {
    ((float*)(ws + WS_PART))[b] = red[0];
    ((int*)(ws + WS_IDX))[b]    = best;
  }
}

// ---------------------------------------------------------------------------
// Scalars: loss = (1+beta)*MSE, perplexity = exp(-sum p log(p+1e-10))
// ---------------------------------------------------------------------------
__global__ __launch_bounds__(256) void finalize_kernel(unsigned* ws, float* out) {
  __shared__ float counts[NE];
  __shared__ float red[256];
  const int tid = threadIdx.x;
  for (int c = tid; c < NE; c += 256) counts[c] = 0.f;
  __syncthreads();
  const int*   idx  = (const int*)(ws + WS_IDX);
  const float* part = (const float*)(ws + WS_PART);
  if (tid < BB) atomicAdd(&counts[idx[tid]], 1.0f);
  red[tid] = (tid < BB) ? part[tid] : 0.f;
  __syncthreads();
  for (int sn = 128; sn > 0; sn >>= 1) { if (tid < sn) red[tid] += red[tid + sn]; __syncthreads(); }
  if (tid == 0) out[65536] = (1.0f + BETA_C) * red[0] / (float)(BB * DD);
  __syncthreads();
  float e = 0.f;
  for (int c = tid; c < NE; c += 256) {
    float p = counts[c] / (float)BB;
    e += p * logf(p + 1e-10f);
  }
  red[tid] = e;
  __syncthreads();
  for (int sn = 128; sn > 0; sn >>= 1) { if (tid < sn) red[tid] += red[tid + sn]; __syncthreads(); }
  if (tid == 0) out[65537] = __expf(-red[0]);
}

extern "C" void kernel_launch(void* const* d_in, const int* in_sizes, int n_in,
                              void* d_out, int out_size, void* d_ws, size_t ws_size,
                              hipStream_t stream) {
  const float* traj = (const float*)d_in[0];
  const int*   mask = (const int*)  d_in[1];
  const float* Wx_f = (const float*)d_in[2];
  const float* Wh_f = (const float*)d_in[3];
  const float* bx_f = (const float*)d_in[4];
  const float* bh_f = (const float*)d_in[5];
  const float* Wx_b = (const float*)d_in[6];
  const float* Wh_b = (const float*)d_in[7];
  const float* bx_b = (const float*)d_in[8];
  const float* bh_b = (const float*)d_in[9];
  const float* cb   = (const float*)d_in[10];
  unsigned* ws = (unsigned*)d_ws;
  float* out = (float*)d_out;

  pack_weights<<<864, 256, 0, stream>>>(Wx_f, Wh_f, Wx_b, Wh_b, ws);
  gru_kernel<<<16, 256, 0, stream>>>(traj, mask, bx_f, bh_f, bx_b, bh_b, ws);
  vq_kernel<<<128, 256, 0, stream>>>(cb, ws, out);
  finalize_kernel<<<1, 256, 0, stream>>>(ws, out);
}
